// Attention_50766513439239
// MI455X (gfx1250) — matrix operand order repaired and checked by reading, not compiled, not run on hardware
//
#include <hip/hip_runtime.h>

// Problem constants (match reference)
#define H   256
#define BB  64
#define S   4096
#define SCHUNK 128      // s columns per workgroup (8 tiles of 16, one per wave)
#define KCHUNK 32       // k depth per staging step
#define NKC    8        // 256 / 32
#define KP     40       // padded bf16 pitch of transposed X tile rows in LDS

typedef __attribute__((ext_vector_type(16))) __bf16 v16bf;
typedef __attribute__((ext_vector_type(8)))  float  v8f;
typedef __attribute__((ext_vector_type(4)))  float  v4f;
typedef __attribute__((ext_vector_type(4)))  int    v4i;

#define AS1 __attribute__((address_space(1)))
#define AS3 __attribute__((address_space(3)))

#if __has_builtin(__builtin_amdgcn_global_load_async_to_lds_b128) && \
    __has_builtin(__builtin_amdgcn_s_wait_asynccnt)
#define USE_ASYNC_LDS 1
#endif

// 16-byte global -> LDS copy; async (ASYNCcnt-tracked) when available.
static __device__ __forceinline__ void copy16_g2l(void* g, void* l) {
#ifdef USE_ASYNC_LDS
  __builtin_amdgcn_global_load_async_to_lds_b128((AS1 v4i*)g, (AS3 v4i*)l, 0, 0);
#else
  *(uint4*)l = *(const uint4*)g;
#endif
}

static __device__ __forceinline__ void wait_async_copies() {
#ifdef USE_ASYNC_LDS
  __builtin_amdgcn_s_wait_asynccnt(0);
#endif
}

static __device__ __forceinline__ unsigned short f32_bf16(float f) {
  unsigned int u = __float_as_uint(f);
  u += 0x7FFFu + ((u >> 16) & 1u);          // round to nearest even
  return (unsigned short)(u >> 16);
}

static __device__ __forceinline__ float fast_tanh(float x) {
#if __has_builtin(__builtin_amdgcn_tanhf)
  return __builtin_amdgcn_tanhf(x);          // V_TANH_F32 (CDNA5 trans op)
#else
  return tanhf(x);
#endif
}

static __device__ __forceinline__ v16bf load_frag(const unsigned short* p) {
  union { uint4 q[2]; v16bf v; } u;          // two ds_load_b128 (16B aligned)
  u.q[0] = *(const uint4*)(p);
  u.q[1] = *(const uint4*)(p + 8);
  return u.v;
}

// ---------------------------------------------------------------------------
// Pre-pack W1/W2 into bf16 WMMA A-fragments (16-bit A 16x32, contiguous k order matching B, not the ISA table order):
// lane L (M = L%16): L<16 holds K = {kc*32 + 0..15},
//                    L>=16 holds K = {kc*32 + 16..31}.
// Fragment order: frag = (kc*2 + m)*16 + ht, 1 KB each (32 lanes x 32 B).
// ---------------------------------------------------------------------------
__global__ void pack_w(const float* __restrict__ W, unsigned short* __restrict__ wsW) {
  int gid  = blockIdx.x * 256 + threadIdx.x;   // 8192 threads total
  int lane = gid & 31;
  int frag = gid >> 5;                          // 0..255
  int ht   = frag & 15;
  int m    = (frag >> 4) & 1;
  int kc   = frag >> 5;
  int h     = ht * 16 + (lane & 15);
  int kbase = kc * 32 + ((lane >= 16) ? 16 : 0);
  const float* src = W + (size_t)h * (3 * H) + m * H;   // W1=W[:, :H], W2=W[:, H:2H]
  unsigned short t[16];
#pragma unroll
  for (int e = 0; e < 16; ++e) {
    int k = kbase + e;
    t[e] = f32_bf16(src[k]);
  }
  unsigned int w[8];
#pragma unroll
  for (int i = 0; i < 8; ++i) w[i] = (unsigned int)t[2*i] | ((unsigned int)t[2*i+1] << 16);
  uint4* dst = (uint4*)(wsW + (size_t)frag * 512 + lane * 16);
  dst[0] = make_uint4(w[0], w[1], w[2], w[3]);
  dst[1] = make_uint4(w[4], w[5], w[6], w[7]);
}

// bias[b,h] = sum_k W3[h,k] * decoder_hidden[b,k]   (8.4 MFLOP total, trivial)
__global__ void bias_k(const float* __restrict__ W, const float* __restrict__ dh,
                       float* __restrict__ bias) {
  int b = blockIdx.x, h = threadIdx.x;
  const float* wrow = W + (size_t)h * (3 * H) + 2 * H;
  const float* d = dh + (size_t)b * H;
  float acc = 0.f;
#pragma unroll 4
  for (int k = 0; k < H; ++k) acc += wrow[k] * d[k];
  bias[(size_t)b * H + h] = acc;
}

// ---------------------------------------------------------------------------
// Main fused kernel: scores[b,s] = sum_h v[h]*tanh( (W1@Xs + W2@Xd)[h,s] + bias[b,h] )
// ---------------------------------------------------------------------------
__launch_bounds__(256, 1)
__global__ void attn_main(const float* __restrict__ Xs, const float* __restrict__ Xd,
                          const unsigned short* __restrict__ wsW,
                          const float* __restrict__ bias, const float* __restrict__ v,
                          float* __restrict__ scores) {
  __shared__ __align__(16) unsigned short ldsA[2 * 16 * 512];     // 32 KB: A-frags for one k-chunk
  __shared__ __align__(16) unsigned short ldsX[2][SCHUNK * KP];   // 20 KB: transposed bf16 X [s][k]
  __shared__ float ldsVB[2 * H];                                   // v | bias(b)

  const int b    = blockIdx.y;
  const int s0   = blockIdx.x * SCHUNK;
  const int tid  = threadIdx.x;
  const int wave = tid >> 5;          // wave owns s-tile `wave`
  const int lane = tid & 31;

  ldsVB[tid]       = v[tid];                    // tid in [0,256)
  ldsVB[H + tid]   = bias[(size_t)b * H + tid];

  v8f acc[16];
  const v8f zero = {0.f, 0.f, 0.f, 0.f, 0.f, 0.f, 0.f, 0.f};
#pragma unroll
  for (int i = 0; i < 16; ++i) acc[i] = zero;

  for (int kc = 0; kc < NKC; ++kc) {
    __syncthreads();   // LDS reuse fence (also covers ldsVB before first use)

    // ---- stage A fragments for this k-chunk: 32 KB from L2-resident packed W.
    //      Async global->LDS (ASYNCcnt) overlaps with the X conversion below.
    {
      unsigned short* src = (unsigned short*)(wsW + (size_t)kc * (2 * 16 * 512));
#pragma unroll
      for (int i = 0; i < 8; ++i) {
        int off = (i * 256 + tid) * 8;           // ushort units, 16 B per copy
        copy16_g2l(src + off, ldsA + off);
      }
    }

    // ---- stage X tiles: stream fp32 (non-temporal b128), cvt->bf16,
    //      transpose to [s][k] with k packed in pairs ----
#pragma unroll
    for (int m = 0; m < 2; ++m) {
      const float* X = (m == 0) ? Xs : Xd;
      const float* xb = X + (size_t)b * H * S + (size_t)(kc * KCHUNK) * S + s0;
      unsigned short* lx = ldsX[m];
#pragma unroll
      for (int it = 0; it < 2; ++it) {          // 512 2x4 micro-tiles / 256 threads
        int e  = it * 256 + tid;
        int sq = e & 31;                         // s quad (coalesced across lanes)
        int kp = e >> 5;                         // k pair 0..15
        v4f r0 = __builtin_nontemporal_load((const v4f*)(xb + (size_t)(kp * 2 + 0) * S + sq * 4));
        v4f r1 = __builtin_nontemporal_load((const v4f*)(xb + (size_t)(kp * 2 + 1) * S + sq * 4));
        int sb = sq * 4;
#pragma unroll
        for (int j = 0; j < 4; ++j) {
          unsigned int pk = (unsigned int)f32_bf16(r0[j]) |
                            ((unsigned int)f32_bf16(r1[j]) << 16);
          *(unsigned int*)(lx + (sb + j) * KP + kp * 2) = pk;
        }
      }
    }
    wait_async_copies();   // this wave's A-frag copies landed in LDS
    __syncthreads();       // everyone's staging visible

    // ---- compute: B-frag per lane = 16 consecutive K for column N=lane%16 ----
    const int scol  = wave * 16 + (lane & 15);
    const int khalf = (lane >> 4) * 16;
    const v16bf bs = load_frag(&ldsX[0][scol * KP + khalf]);
    const v16bf bd = load_frag(&ldsX[1][scol * KP + khalf]);
#pragma unroll
    for (int ht = 0; ht < 16; ++ht) {
      const v16bf a1 = load_frag(&ldsA[(0 * 16 + ht) * 512 + lane * 16]);
      const v16bf a2 = load_frag(&ldsA[(1 * 16 + ht) * 512 + lane * 16]);
      acc[ht] = __builtin_amdgcn_wmma_f32_16x16x32_bf16(false, a1, false, bs,
                                                        (short)0, acc[ht], false, false);
      acc[ht] = __builtin_amdgcn_wmma_f32_16x16x32_bf16(false, a2, false, bd,
                                                        (short)0, acc[ht], false, false);
    }
  }

  // ---- epilogue: h-reduction of v[h]*tanh(e + bias) (C layout: VGPR j ->
  // M = ht*16 + j (+8 for hi lanes), N = lane%16), then half-wave sum ----
  float p = 0.f;
#pragma unroll
  for (int ht = 0; ht < 16; ++ht) {
    int hbase = ht * 16 + ((lane >= 16) ? 8 : 0);
#pragma unroll
    for (int j = 0; j < 8; ++j) {
      float e = acc[ht][j] + ldsVB[H + hbase + j];
      p += ldsVB[hbase + j] * fast_tanh(e);
    }
  }
  p += __shfl_xor(p, 16, 32);
  if (lane < 16)
    scores[(size_t)b * S + s0 + wave * 16 + lane] = p;
}

// In-place row softmax over S=4096 (d_out fully rewritten by attn_main first)
__global__ void softmax_k(float* __restrict__ out) {
  __shared__ float smax[8], ssum[8];
  int b = blockIdx.x, tid = threadIdx.x;
  float* row = out + (size_t)b * S;
  float vals[16];
  float mx = -3.4e38f;
#pragma unroll
  for (int i = 0; i < 16; ++i) { vals[i] = row[tid + i * 256]; mx = fmaxf(mx, vals[i]); }
#pragma unroll
  for (int off = 16; off > 0; off >>= 1) mx = fmaxf(mx, __shfl_xor(mx, off, 32));
  if ((tid & 31) == 0) smax[tid >> 5] = mx;
  __syncthreads();
  float bmax = smax[0];
#pragma unroll
  for (int i = 1; i < 8; ++i) bmax = fmaxf(bmax, smax[i]);
  float s = 0.f;
#pragma unroll
  for (int i = 0; i < 16; ++i) { vals[i] = __expf(vals[i] - bmax); s += vals[i]; }
#pragma unroll
  for (int off = 16; off > 0; off >>= 1) s += __shfl_xor(s, off, 32);
  if ((tid & 31) == 0) ssum[tid >> 5] = s;
  __syncthreads();
  float tot = 0.f;
#pragma unroll
  for (int i = 0; i < 8; ++i) tot += ssum[i];
  float inv = 1.f / tot;
#pragma unroll
  for (int i = 0; i < 16; ++i) row[tid + i * 256] = vals[i] * inv;
}

extern "C" void kernel_launch(void* const* d_in, const int* in_sizes, int n_in,
                              void* d_out, int out_size, void* d_ws, size_t ws_size,
                              hipStream_t stream) {
  const float* static_enc     = (const float*)d_in[0];   // (B,H,S)
  const float* dynamic_enc    = (const float*)d_in[1];   // (B,H,S)
  const float* decoder_hidden = (const float*)d_in[2];   // (B,H)
  const float* v              = (const float*)d_in[3];   // (1,H)
  const float* W              = (const float*)d_in[4];   // (H,3H)

  unsigned short* wsW  = (unsigned short*)d_ws;              // 256 KB A-fragments
  float*          bias = (float*)((char*)d_ws + 262144);     // 64 KB bias[b,h]
  float*          out  = (float*)d_out;

  pack_w<<<32, 256, 0, stream>>>(W, wsW);
  bias_k<<<BB, 256, 0, stream>>>(W, decoder_hidden, bias);
  dim3 grid(S / SCHUNK, BB);
  attn_main<<<grid, 256, 0, stream>>>(static_enc, dynamic_enc, wsW, bias, v, out);
  softmax_k<<<BB, 256, 0, stream>>>(out);
}
